// TransformerEncoder_62612033241691
// MI455X (gfx1250) — compile-verified, matrix operand order checked by reading, not run on hardware
//
#include <hip/hip_runtime.h>
#include <hip/hip_bf16.h>
#include <math.h>

// ---------------------------------------------------------------------------
// Types for CDNA5 WMMA (wave32): v_wmma_f32_16x16x32_bf16
// ---------------------------------------------------------------------------
typedef __attribute__((ext_vector_type(16))) __bf16 v16bf;
typedef __attribute__((ext_vector_type(8)))  __bf16 v8bf;
typedef __attribute__((ext_vector_type(8)))  float  v8f;

#define FEATS 768
#define HEADS 12
#define DH 64
#define SEQ 512
#define BATCH 16
#define MROWS (BATCH * SEQ)        // 8192
#define MLPH 3072
#define LN_EPS 1e-5f
#define ATTN_SCALE 0.0360843918f   // 1/sqrt(768)

static __device__ __forceinline__ v8f wmma_bf16(v16bf a, v16bf b, v8f c) {
    // D = A(16x32 bf16) * B(32x16 bf16) + C(16x16 f32)
    return __builtin_amdgcn_wmma_f32_16x16x32_bf16(
        /*neg_a=*/false, a, /*neg_b=*/false, b,
        /*c_mod=*/(short)0, c, /*reuse_a=*/false, /*reuse_b=*/false);
}

// Load a 16x32 bf16 A-style fragment (also used for B fragments, since the
// ISA layouts are symmetric: lane = row/col index, K split in 8-wide chunks).
// base: row-major bf16 with leading dim `ld` (elements), rows = lane&15,
// lanes<16 hold K {0..7,16..23}, lanes>=16 hold K {8..15,24..31}.
static __device__ __forceinline__ v16bf load_frag(const __bf16* base, int ld, int lane) {
    const int r  = lane & 15;
    const int kh = (lane >> 4) << 3;        // 0 or 8
    v8bf lo = *(const v8bf*)(base + (size_t)r * ld + kh);
    v8bf hi = *(const v8bf*)(base + (size_t)r * ld + 16 + kh);
    v16bf out;
#pragma unroll
    for (int i = 0; i < 8; ++i) { out[i] = lo[i]; out[i + 8] = hi[i]; }
    return out;
}

static __device__ __forceinline__ float gelu_exact(float v) {
    return 0.5f * v * (1.0f + erff(v * 0.70710678118f));
}

// ---------------------------------------------------------------------------
// LayerNorm (f32 in) -> bf16 out.  One block per row of 768, 256 threads.
// ---------------------------------------------------------------------------
__global__ void ln_bf16_kernel(const float* __restrict__ x,
                               const float* __restrict__ w,
                               const float* __restrict__ b,
                               __bf16* __restrict__ out) {
    const int row = blockIdx.x;
    const int t = threadIdx.x;
    const int lane = t & 31, wv = t >> 5;
    const float* xr = x + (size_t)row * FEATS;

    float a0 = xr[t], a1 = xr[t + 256], a2 = xr[t + 512];
    float s = a0 + a1 + a2;
#pragma unroll
    for (int m = 16; m >= 1; m >>= 1) s += __shfl_xor(s, m, 32);
    __shared__ float sb[8];
    if (lane == 0) sb[wv] = s;
    __syncthreads();
    float tot = 0.f;
#pragma unroll
    for (int i = 0; i < 8; ++i) tot += sb[i];
    const float mean = tot * (1.0f / FEATS);

    float d0 = a0 - mean, d1 = a1 - mean, d2 = a2 - mean;
    float q = d0 * d0 + d1 * d1 + d2 * d2;
#pragma unroll
    for (int m = 16; m >= 1; m >>= 1) q += __shfl_xor(q, m, 32);
    __syncthreads();
    if (lane == 0) sb[wv] = q;
    __syncthreads();
    float tot2 = 0.f;
#pragma unroll
    for (int i = 0; i < 8; ++i) tot2 += sb[i];
    const float rstd = rsqrtf(tot2 * (1.0f / FEATS) + LN_EPS);

    __bf16* orow = out + (size_t)row * FEATS;
    orow[t]       = (__bf16)(d0 * rstd * w[t]       + b[t]);
    orow[t + 256] = (__bf16)(d1 * rstd * w[t + 256] + b[t + 256]);
    orow[t + 512] = (__bf16)(d2 * rstd * w[t + 512] + b[t + 512]);
}

// ---------------------------------------------------------------------------
// Tiled WMMA GEMM:  out = epilogue(A[MxK](bf16) * W[KxN](f32) + bias)
// Block tile 128x128, BK=32, 128 threads (4 waves in 2x2), 64x64 per wave
// (4x4 fragments -> 16 WMMA per K-step, 1.0 WMMA per ds_load_b128).
// Double-buffered LDS so tile k+1 global loads overlap tile k WMMAs.
// EPI: 0 = QKV head-scatter (bf16 -> [B,H,N,64])
//      1 = GELU, bf16 out [MxN]
//      2 = f32 out = val*gamma[col] + res[row,col]
//      3 = f32 out = gelu(val)*gamma[col] + res[row,col]
// ---------------------------------------------------------------------------
#define LDA_PAD 40   // bf16 elements per LDS row (32 data + pad), 80B rows
template <int EPI>
__global__ void gemm_bf16_kernel(const __bf16* __restrict__ A,
                                 const float* __restrict__ W,
                                 const float* __restrict__ bias,
                                 const float* __restrict__ gamma,
                                 const float* __restrict__ res,
                                 void* __restrict__ outp,
                                 int M, int K, int Nn) {
    __shared__ __bf16 As[2][128 * LDA_PAD];
    __shared__ __bf16 Bs[2][128 * LDA_PAD];   // transposed: Bs[n][k]

    const int t = threadIdx.x;
    const int lane = t & 31, wid = t >> 5;
    const int wm = wid >> 1, wn = wid & 1;           // 2x2 wave grid
    const int nB = blockIdx.x * 128;
    const int mB = blockIdx.y * 128;

    v8f acc[4][4];
#pragma unroll
    for (int i = 0; i < 4; ++i)
#pragma unroll
        for (int j = 0; j < 4; ++j) acc[i][j] = (v8f){};

    const int nK = K >> 5;

    // Stage one 128x32 A tile + 32x128 W tile (transposed, f32->bf16) into buf.
    auto stage = [&](int kt, int buf) {
        const int kb = kt << 5;
#pragma unroll
        for (int it = 0; it < 4; ++it) {
            const int c = t + it * 128;              // 512 chunks of 8 bf16
            const int row = c >> 2, kc = c & 3;
            *(v8bf*)&As[buf][row * LDA_PAD + kc * 8] =
                *(const v8bf*)&A[(size_t)(mB + row) * K + kb + kc * 8];
        }
        const int kcol = t & 31;                     // k within tile
#pragma unroll
        for (int g = 0; g < 2; ++g) {
            const int ng = (t >> 5) + g * 4;         // 8 groups of 16 n
            const float* wp = W + (size_t)(kb + kcol) * Nn + nB + ng * 16;
#pragma unroll
            for (int j4 = 0; j4 < 4; ++j4) {
                float4 f = *(const float4*)(wp + j4 * 4);
                const int n0 = ng * 16 + j4 * 4;
                Bs[buf][(n0 + 0) * LDA_PAD + kcol] = (__bf16)f.x;
                Bs[buf][(n0 + 1) * LDA_PAD + kcol] = (__bf16)f.y;
                Bs[buf][(n0 + 2) * LDA_PAD + kcol] = (__bf16)f.z;
                Bs[buf][(n0 + 3) * LDA_PAD + kcol] = (__bf16)f.w;
            }
        }
    };

    stage(0, 0);
    int cur = 0;
    for (int kt = 0; kt < nK; ++kt) {
        __syncthreads();                 // buf[cur] ready; prev reads of cur^1 done
        if (kt + 1 < nK) stage(kt + 1, cur ^ 1);

        v16bf a[4], b[4];
#pragma unroll
        for (int i = 0; i < 4; ++i) {
            a[i] = load_frag(As[cur] + (wm * 64 + i * 16) * LDA_PAD, LDA_PAD, lane);
            b[i] = load_frag(Bs[cur] + (wn * 64 + i * 16) * LDA_PAD, LDA_PAD, lane);
        }
#pragma unroll
        for (int fr = 0; fr < 4; ++fr)
#pragma unroll
            for (int fc = 0; fc < 4; ++fc)
                acc[fr][fc] = wmma_bf16(a[fr], b[fc], acc[fr][fc]);
        cur ^= 1;
    }

    // Epilogue.  C/D layout: lane&15 -> column, vgpr v -> row v (+8 if lane>=16).
    const int hl = lane >> 4, cl = lane & 15;
#pragma unroll
    for (int fr = 0; fr < 4; ++fr)
#pragma unroll
        for (int fc = 0; fc < 4; ++fc)
#pragma unroll
            for (int v = 0; v < 8; ++v) {
                const int row = mB + wm * 64 + fr * 16 + v + 8 * hl;
                const int col = nB + wn * 64 + fc * 16 + cl;
                float val = acc[fr][fc][v] + bias[col];
                if constexpr (EPI == 1 || EPI == 3) val = gelu_exact(val);
                if constexpr (EPI == 0) {
                    const int bb = row >> 9, ii = row & 511;
                    const int hh = col >> 6, dd = col & 63;
                    ((__bf16*)outp)[(((size_t)(bb * HEADS + hh) * SEQ) + ii) * DH + dd] =
                        (__bf16)val;
                } else if constexpr (EPI == 1) {
                    ((__bf16*)outp)[(size_t)row * Nn + col] = (__bf16)val;
                } else {
                    float r2 = val * gamma[col] + res[(size_t)row * FEATS + col];
                    ((float*)outp)[(size_t)row * FEATS + col] = r2;
                }
            }
}

// ---------------------------------------------------------------------------
// Flash attention: one block per (b*h, 64-query tile).  128 threads, 4 waves,
// each wave owns 16 query rows.  Q/K/V are bf16 [B*H, 512, 64].
// ---------------------------------------------------------------------------
#define SP 72   // padded LDS row (64 data + 8), 144B rows
__global__ void attn_kernel(const __bf16* __restrict__ Q,
                            const __bf16* __restrict__ K,
                            const __bf16* __restrict__ V,
                            __bf16* __restrict__ out) {
    __shared__ __bf16 Qs[64 * SP];
    __shared__ __bf16 Ks[64 * SP];
    __shared__ __bf16 Vt[64 * SP];            // transposed: Vt[d][j]
    __shared__ __bf16 Ps[4 * 16 * SP];        // per-wave P tiles

    const int bh = blockIdx.y;
    const int qt = blockIdx.x;
    const int t = threadIdx.x;
    const int lane = t & 31, wid = t >> 5;
    const int hl = lane >> 4, cl = lane & 15;

    // Load Q tile (64x64 bf16) with 16B vector copies.
    const __bf16* Qg = Q + (size_t)(bh * SEQ + qt * 64) * DH;
#pragma unroll
    for (int it = 0; it < 4; ++it) {
        const int idx = t + it * 128;         // 512 chunks of 8 bf16
        const int row = idx >> 3, ch = idx & 7;
        *(v8bf*)&Qs[row * SP + ch * 8] = *(const v8bf*)&Qg[row * DH + ch * 8];
    }

    v8f o[4];
#pragma unroll
    for (int i = 0; i < 4; ++i) o[i] = (v8f){};
    float mstat[8], lstat[8];
#pragma unroll
    for (int v = 0; v < 8; ++v) { mstat[v] = -3.0e38f; lstat[v] = 0.f; }

    for (int jt = 0; jt < SEQ / 64; ++jt) {
        __syncthreads();
        const __bf16* Kg = K + (size_t)(bh * SEQ + jt * 64) * DH;
        const __bf16* Vg = V + (size_t)(bh * SEQ + jt * 64) * DH;
#pragma unroll
        for (int it = 0; it < 4; ++it) {
            const int idx = t + it * 128;
            const int row = idx >> 3, ch = idx & 7;
            *(v8bf*)&Ks[row * SP + ch * 8] = *(const v8bf*)&Kg[row * DH + ch * 8];
            v8bf vv = *(const v8bf*)&Vg[row * DH + ch * 8];
#pragma unroll
            for (int e = 0; e < 8; ++e) Vt[(ch * 8 + e) * SP + row] = vv[e];
        }
        __syncthreads();

        // S = Q * K^T (16x64 per wave), scaled.
        v8f s[4];
#pragma unroll
        for (int jj = 0; jj < 4; ++jj) s[jj] = (v8f){};
#pragma unroll
        for (int kk = 0; kk < 2; ++kk) {
            v16bf a = load_frag(Qs + (wid * 16) * SP + kk * 32, SP, lane);
#pragma unroll
            for (int jj = 0; jj < 4; ++jj) {
                v16bf b = load_frag(Ks + (jj * 16) * SP + kk * 32, SP, lane);
                s[jj] = wmma_bf16(a, b, s[jj]);
            }
        }
#pragma unroll
        for (int jj = 0; jj < 4; ++jj)
#pragma unroll
            for (int v = 0; v < 8; ++v) s[jj][v] *= ATTN_SCALE;

        // Online softmax row stats (row = v + 8*hl, spread over 16 lanes).
        float rm[8];
#pragma unroll
        for (int v = 0; v < 8; ++v) {
            rm[v] = fmaxf(fmaxf(s[0][v], s[1][v]), fmaxf(s[2][v], s[3][v]));
#pragma unroll
            for (int m = 8; m >= 1; m >>= 1) rm[v] = fmaxf(rm[v], __shfl_xor(rm[v], m, 16));
        }
        float alpha[8], rs[8];
#pragma unroll
        for (int v = 0; v < 8; ++v) {
            const float mnew = fmaxf(mstat[v], rm[v]);
            alpha[v] = __expf(mstat[v] - mnew);
            mstat[v] = mnew;
            rs[v] = 0.f;
#pragma unroll
            for (int jj = 0; jj < 4; ++jj) {
                float p = __expf(s[jj][v] - mnew);
                s[jj][v] = p;
                rs[v] += p;
            }
#pragma unroll
            for (int m = 8; m >= 1; m >>= 1) rs[v] += __shfl_xor(rs[v], m, 16);
            lstat[v] = lstat[v] * alpha[v] + rs[v];
        }
#pragma unroll
        for (int dd = 0; dd < 4; ++dd)
#pragma unroll
            for (int v = 0; v < 8; ++v) o[dd][v] *= alpha[v];

        // Re-fragment P through per-wave LDS (wave-local: LDS is in-order).
#pragma unroll
        for (int jj = 0; jj < 4; ++jj)
#pragma unroll
            for (int v = 0; v < 8; ++v)
                Ps[(wid * 16 + v + 8 * hl) * SP + jj * 16 + cl] = (__bf16)s[jj][v];
        asm volatile("s_wait_dscnt 0" ::: "memory");

        // O += P * V
#pragma unroll
        for (int kk = 0; kk < 2; ++kk) {
            v16bf a = load_frag(Ps + (wid * 16) * SP + kk * 32, SP, lane);
#pragma unroll
            for (int dd = 0; dd < 4; ++dd) {
                v16bf b = load_frag(Vt + (dd * 16) * SP + kk * 32, SP, lane);
                o[dd] = wmma_bf16(a, b, o[dd]);
            }
        }
    }

    // Normalize and write [B, N, F] bf16 (heads concatenated).
    const int bb = bh / HEADS, hh = bh % HEADS;
#pragma unroll
    for (int v = 0; v < 8; ++v) {
        const float inv = 1.0f / lstat[v];
        const int i = qt * 64 + wid * 16 + v + 8 * hl;
#pragma unroll
        for (int dd = 0; dd < 4; ++dd) {
            const int col = hh * DH + dd * 16 + cl;
            out[((size_t)(bb * SEQ + i)) * FEATS + col] = (__bf16)(o[dd][v] * inv);
        }
    }
}

// ---------------------------------------------------------------------------
// Host-side orchestration
// ---------------------------------------------------------------------------
extern "C" void kernel_launch(void* const* d_in, const int* in_sizes, int n_in,
                              void* d_out, int out_size, void* d_ws, size_t ws_size,
                              hipStream_t stream) {
    const float* x      = (const float*)d_in[0];
    const float* ln1_w  = (const float*)d_in[1];
    const float* ln1_b  = (const float*)d_in[2];
    const float* wq     = (const float*)d_in[3];
    const float* bq     = (const float*)d_in[4];
    const float* wk     = (const float*)d_in[5];
    const float* bk     = (const float*)d_in[6];
    const float* wv     = (const float*)d_in[7];
    const float* bv     = (const float*)d_in[8];
    const float* wo     = (const float*)d_in[9];
    const float* bo     = (const float*)d_in[10];
    const float* gamma1 = (const float*)d_in[11];
    const float* ln2_w  = (const float*)d_in[12];
    const float* ln2_b  = (const float*)d_in[13];
    const float* w1     = (const float*)d_in[14];
    const float* b1     = (const float*)d_in[15];
    const float* w2     = (const float*)d_in[16];
    const float* b2     = (const float*)d_in[17];
    const float* gamma2 = (const float*)d_in[18];
    float* out          = (float*)d_out;

    // Workspace layout (bytes).  attnOut aliases xn; yn aliases Q.
    char* w = (char*)d_ws;
    const size_t S1 = (size_t)MROWS * FEATS * sizeof(__bf16);   // 12.58 MB
    __bf16* xn   = (__bf16*)(w);                 // also attnOut
    __bf16* qb   = (__bf16*)(w + S1);            // also yn
    __bf16* kb   = (__bf16*)(w + 2 * S1);
    __bf16* vb   = (__bf16*)(w + 3 * S1);
    float*  out1 = (float*) (w + 4 * S1);        // 25.17 MB
    __bf16* hdn  = (__bf16*)(w + 4 * S1 + (size_t)MROWS * FEATS * sizeof(float));

    const dim3 blk(128);
    const dim3 g768(FEATS / 128, MROWS / 128);    // (6, 64)
    const dim3 g3072(MLPH / 128, MROWS / 128);    // (24, 64)

    // 1) LN1 -> bf16 xn
    ln_bf16_kernel<<<MROWS, 256, 0, stream>>>(x, ln1_w, ln1_b, xn);

    // 2) QKV projections with head scatter -> [B,H,N,64] bf16
    gemm_bf16_kernel<0><<<g768, blk, 0, stream>>>(xn, wq, bq, nullptr, nullptr, qb, MROWS, FEATS, FEATS);
    gemm_bf16_kernel<0><<<g768, blk, 0, stream>>>(xn, wk, bk, nullptr, nullptr, kb, MROWS, FEATS, FEATS);
    gemm_bf16_kernel<0><<<g768, blk, 0, stream>>>(xn, wv, bv, nullptr, nullptr, vb, MROWS, FEATS, FEATS);

    // 3) Flash attention -> attnOut (aliases xn) [B,N,F] bf16
    attn_kernel<<<dim3(SEQ / 64, BATCH * HEADS), blk, 0, stream>>>(qb, kb, vb, xn);

    // 4) Output projection + gamma1 * (.) + x  ->  out1 (f32)
    gemm_bf16_kernel<2><<<g768, blk, 0, stream>>>(xn, wo, bo, gamma1, x, out1, MROWS, FEATS, FEATS);

    // 5) LN2 -> bf16 yn (aliases Q)
    ln_bf16_kernel<<<MROWS, 256, 0, stream>>>(out1, ln2_w, ln2_b, qb);

    // 6) MLP up + exact GELU -> hdn bf16 [M, 3072]
    gemm_bf16_kernel<1><<<g3072, blk, 0, stream>>>(qb, w1, b1, nullptr, nullptr, hdn, MROWS, FEATS, MLPH);

    // 7) MLP down + exact GELU + gamma2 * (.) + out1 -> d_out (f32)
    gemm_bf16_kernel<3><<<g768, blk, 0, stream>>>(hdn, w2, b2, gamma2, out1, out, MROWS, MLPH, FEATS);
}
